// CausalSelfAttention_18502719111656
// MI455X (gfx1250) — matrix operand order repaired and checked by reading, not compiled, not run on hardware
//
#include <hip/hip_runtime.h>
#include <cstdint>

// Causal self-attention fwd for B=2, T=2048, C=768, H=12, hs=64 on gfx1250.
// Pipeline: (1) QKV GEMM fp32->bf16 with v_wmma_f32_16x16x32_bf16,
//           (2) flash-attention (streaming softmax, WMMA for S and P@V,
//               double-buffered GLOBAL_LOAD_ASYNC_TO_LDS_B128 staging),
//           (3) output projection GEMM -> fp32 (double-buffered async A tile).
// Workspace layout (bf16): Kh | Qh | Vt | Y  (4 x 6.29 MB = ~25.2 MB).

#define B_DIM 2
#define T_DIM 2048
#define C_DIM 768
#define H_DIM 12
#define HS_DIM 64
#define M_DIM (B_DIM * T_DIM) /* 4096 */

typedef __bf16 bf16;
typedef bf16 bf16x4 __attribute__((ext_vector_type(4)));
typedef bf16 bf16x8 __attribute__((ext_vector_type(8)));
typedef bf16 v16bf __attribute__((ext_vector_type(16)));
typedef float v8f __attribute__((ext_vector_type(8)));

union ABFrag {
  v16bf v;
  bf16x8 h[2];
};

static __device__ __forceinline__ v8f wmma_bf16(v16bf a, v16bf b, v8f c) {
  // 8 args: (neg_a, A, neg_b, B, c_mod, C, reuse_a, reuse_b)
  return __builtin_amdgcn_wmma_f32_16x16x32_bf16(false, a, false, b, (short)0,
                                                 c, false, false);
}

// Async global->LDS 16B copy (CDNA5, ASYNCcnt-tracked, no VGPR round-trip).
// lds_off: wave-relative LDS byte offset (low 32 bits of the flat pointer ==
// AS(3) address per the LDS aperture mapping). gaddr: 64-bit global address.
static __device__ __forceinline__ void async_copy_b128(unsigned lds_off,
                                                       const void* gaddr) {
  asm volatile("global_load_async_to_lds_b128 %0, %1, off" ::"v"(lds_off),
               "v"(gaddr)
               : "memory");
}
static __device__ __forceinline__ void async_wait0() {
  asm volatile("s_wait_asynccnt 0x0" ::: "memory");
}
// Async loads complete in order: with 2 newer transfers outstanding, waiting
// ASYNCcnt<=2 guarantees the 2 older (current-tile) transfers have landed.
static __device__ __forceinline__ void async_wait_le2() {
  asm volatile("s_wait_asynccnt 0x2" ::: "memory");
}
static __device__ __forceinline__ unsigned lds_off(const void* p) {
  return (unsigned)(uintptr_t)p;
}

// ---------------------------------------------------------------------------
// Kernel 1: fused QKV projection. grid = (N/64, M/128, 3), block = 256 (8 waves)
// z=0 -> Wk/bk -> Kh[b][h][t][s]; z=1 -> Wq/bq -> Qh[b][h][t][s];
// z=2 -> Wv/bv -> Vt[b][h][s][t] (transposed for contiguous-K B fragments).
// A/B staging stays on the VGPR path (fp32->bf16 conversion is required);
// the compiler software-pipelines those global loads across the barrier.
// ---------------------------------------------------------------------------
__global__ __launch_bounds__(256) void qkv_proj_kernel(
    const float* __restrict__ x, const float* __restrict__ Wk,
    const float* __restrict__ bk, const float* __restrict__ Wq,
    const float* __restrict__ bq, const float* __restrict__ Wv,
    const float* __restrict__ bv, bf16* __restrict__ Kh, bf16* __restrict__ Qh,
    bf16* __restrict__ Vt) {
  const int z = blockIdx.z;
  const float* W = (z == 0) ? Wk : (z == 1) ? Wq : Wv;
  const float* bi = (z == 0) ? bk : (z == 1) ? bq : bv;

  const int m0 = blockIdx.y * 128;
  const int n0 = blockIdx.x * 64;
  const int tid = threadIdx.x;
  const int lane = tid & 31;
  const int wave = tid >> 5;
  const int wm = wave & 3;   // 4 strips of 32 rows
  const int wn = wave >> 2;  // 2 strips of 32 cols
  const int ln = lane & 15;
  const int half = lane >> 4;

  __shared__ __align__(16) bf16 As[128 * 40];  // x tile, row-major [m][k]
  __shared__ __align__(16) bf16 Bs[64 * 40];   // W tile transposed [n][k]

  const v8f vzero = {0.f, 0.f, 0.f, 0.f, 0.f, 0.f, 0.f, 0.f};
  v8f acc[2][2] = {{vzero, vzero}, {vzero, vzero}};

  for (int k0 = 0; k0 < C_DIM; k0 += 32) {
    __syncthreads();
#pragma unroll
    for (int p = 0; p < 4; ++p) {  // 128x32 f32 -> bf16 (conversion needed)
      int linear = (p * 256 + tid) * 4;
      int r = linear >> 5, cc = linear & 31;
      const float4 f = *(const float4*)(x + (size_t)(m0 + r) * C_DIM + k0 + cc);
      bf16x4 o = {(bf16)f.x, (bf16)f.y, (bf16)f.z, (bf16)f.w};
      *(bf16x4*)(As + r * 40 + cc) = o;
    }
    // W tile: 128 threads each handle a 4x4 patch, packed b64 stores into [n][k]
    if (tid < 128) {
      const int ng = (tid & 15) * 4;
      const int kg = (tid >> 4) * 4;
      const float* wp = W + (size_t)(k0 + kg) * C_DIM + n0 + ng;
      const float4 f0 = *(const float4*)(wp);
      const float4 f1 = *(const float4*)(wp + C_DIM);
      const float4 f2 = *(const float4*)(wp + 2 * C_DIM);
      const float4 f3 = *(const float4*)(wp + 3 * C_DIM);
      bf16x4 c0 = {(bf16)f0.x, (bf16)f1.x, (bf16)f2.x, (bf16)f3.x};
      bf16x4 c1 = {(bf16)f0.y, (bf16)f1.y, (bf16)f2.y, (bf16)f3.y};
      bf16x4 c2 = {(bf16)f0.z, (bf16)f1.z, (bf16)f2.z, (bf16)f3.z};
      bf16x4 c3 = {(bf16)f0.w, (bf16)f1.w, (bf16)f2.w, (bf16)f3.w};
      *(bf16x4*)(Bs + (ng + 0) * 40 + kg) = c0;
      *(bf16x4*)(Bs + (ng + 1) * 40 + kg) = c1;
      *(bf16x4*)(Bs + (ng + 2) * 40 + kg) = c2;
      *(bf16x4*)(Bs + (ng + 3) * 40 + kg) = c3;
    }
    __syncthreads();

    ABFrag af[2], bfm[2];
#pragma unroll
    for (int mt = 0; mt < 2; ++mt) {
      int row = wm * 32 + mt * 16 + ln;
      int kb = half ? 16 : 0;  // A layout: lanes 0-15 K0..15, 16-31 K16..31
      af[mt].h[0] = *(const bf16x8*)(As + row * 40 + kb);
      af[mt].h[1] = *(const bf16x8*)(As + row * 40 + kb + 8);
    }
#pragma unroll
    for (int nt = 0; nt < 2; ++nt) {
      int nrow = wn * 32 + nt * 16 + ln;
      int kb = half ? 16 : 0;  // B layout: lanes 0-15 K0..15, 16-31 K16..31
      bfm[nt].h[0] = *(const bf16x8*)(Bs + nrow * 40 + kb);
      bfm[nt].h[1] = *(const bf16x8*)(Bs + nrow * 40 + kb + 8);
    }
#pragma unroll
    for (int mt = 0; mt < 2; ++mt)
#pragma unroll
      for (int nt = 0; nt < 2; ++nt)
        acc[mt][nt] = wmma_bf16(af[mt].v, bfm[nt].v, acc[mt][nt]);
  }

#pragma unroll
  for (int nt = 0; nt < 2; ++nt) {
    int n = n0 + wn * 32 + nt * 16 + ln;
    float bias = bi[n];
    int hh = n >> 6, ss = n & 63;
#pragma unroll
    for (int mt = 0; mt < 2; ++mt) {
#pragma unroll
      for (int r = 0; r < 8; ++r) {
        int m = m0 + wm * 32 + mt * 16 + half * 8 + r;
        float val = acc[mt][nt][r] + bias;
        int bb = m >> 11, tt = m & (T_DIM - 1);
        if (z == 2) {
          Vt[((size_t)(bb * H_DIM + hh) * HS_DIM + ss) * T_DIM + tt] =
              (bf16)val;
        } else {
          bf16* dst = (z == 0) ? Kh : Qh;
          dst[((size_t)(bb * H_DIM + hh) * T_DIM + tt) * HS_DIM + ss] =
              (bf16)val;
        }
      }
    }
  }
}

// ---------------------------------------------------------------------------
// Kernel 2: flash attention. att[t,j] = k_t . q_j / 8, causal j<=t, softmax,
// y_t = sum_j p[t,j] v_j.  grid = (T/128, H, B), block = 256 (8 waves x 16
// rows). Q/V tiles are double-buffered: next tile streams in via async DMA
// while WMMA + softmax run on the current tile.
// ---------------------------------------------------------------------------
__global__ __launch_bounds__(256) void flash_attn_kernel(
    const bf16* __restrict__ Kh, const bf16* __restrict__ Qh,
    const bf16* __restrict__ Vt, bf16* __restrict__ Y) {
  const int t0 = blockIdx.x * 128;
  const int h = blockIdx.y;
  const int b = blockIdx.z;
  const int bh = b * H_DIM + h;
  const int tid = threadIdx.x;
  const int lane = tid & 31;
  const int wave = tid >> 5;
  const int ln = lane & 15;
  const int half = lane >> 4;

  __shared__ __align__(16) bf16 Qs[2][32 * 72];  // [buf][j_local][s]
  __shared__ __align__(16) bf16 Vs[2][64 * 40];  // [buf][s'][i_local]
  __shared__ __align__(16) bf16 Ps[8][16 * 40];  // wave-private P staging

  const bf16* Kbase = Kh + (size_t)bh * T_DIM * HS_DIM;
  const bf16* Qbase = Qh + (size_t)bh * T_DIM * HS_DIM;
  const bf16* Vbase = Vt + (size_t)bh * HS_DIM * T_DIM;

  // Per-wave K-row A-fragments (rows t0+wave*16 .. +15), kept for whole sweep.
  ABFrag ak[2];
  {
    int trow = t0 + wave * 16 + ln;
#pragma unroll
    for (int ks = 0; ks < 2; ++ks) {
      int sb = ks * 32 + (half ? 16 : 0);
      ak[ks].h[0] = *(const bf16x8*)(Kbase + (size_t)trow * HS_DIM + sb);
      ak[ks].h[1] = *(const bf16x8*)(Kbase + (size_t)trow * HS_DIM + sb + 8);
    }
  }

  const v8f vzero = {0.f, 0.f, 0.f, 0.f, 0.f, 0.f, 0.f, 0.f};
  v8f O[4] = {vzero, vzero, vzero, vzero};
  float mrow[8], lrow[8];
#pragma unroll
  for (int r = 0; r < 8; ++r) {
    mrow[r] = -1e30f;
    lrow[r] = 0.f;
  }

  // Per-thread async staging addresses (fixed across the j sweep).
  const int qr = (tid * 8) >> 6, qc = (tid * 8) & 63;
  const int vr = (tid * 8) >> 5, vc = (tid * 8) & 31;
  const unsigned q_dst[2] = {lds_off(Qs[0] + qr * 72 + qc),
                             lds_off(Qs[1] + qr * 72 + qc)};
  const unsigned v_dst[2] = {lds_off(Vs[0] + vr * 40 + vc),
                             lds_off(Vs[1] + vr * 40 + vc)};

  const int rbase = t0 + wave * 16 + half * 8;
  const float sc = 0.125f * 1.4426950408889634f;  // 1/sqrt(64) * log2(e)
  const int wave_row_max = t0 + wave * 16 + 15;
  const int ntiles = (t0 + 128) / 32;  // columns 0 .. t0+127

  // Prologue: kick off tile 0 into buffer 0.
  async_copy_b128(q_dst[0], Qbase + (size_t)qr * HS_DIM + qc);
  async_copy_b128(v_dst[0], Vbase + (size_t)vr * T_DIM + vc);

  for (int it = 0; it < ntiles; ++it) {
    const int j0 = it * 32;
    const int cur = it & 1;
    __syncthreads();  // all waves done reading buf[cur^1] (iteration it-1)
    if (it + 1 < ntiles) {
      const int jn = j0 + 32;
      async_copy_b128(q_dst[cur ^ 1], Qbase + (size_t)(jn + qr) * HS_DIM + qc);
      async_copy_b128(v_dst[cur ^ 1], Vbase + (size_t)vr * T_DIM + jn + vc);
      async_wait_le2();  // tile `it` landed; tile `it+1` still in flight
    } else {
      async_wait0();
    }
    __syncthreads();  // tile `it` visible to every wave

    if (j0 <= wave_row_max) {  // skip tiles fully above the diagonal
      const bf16* Qc = Qs[cur];
      const bf16* Vc = Vs[cur];

      // S (16 rows x 32 cols) = K_rows x Q^T
      v8f S[2];
#pragma unroll
      for (int jt = 0; jt < 2; ++jt) {
        v8f s = vzero;
#pragma unroll
        for (int ks = 0; ks < 2; ++ks) {
          ABFrag bq;
          int nrow = jt * 16 + ln;
          int kb = ks * 32 + (half ? 16 : 0);
          bq.h[0] = *(const bf16x8*)(Qc + nrow * 72 + kb);
          bq.h[1] = *(const bf16x8*)(Qc + nrow * 72 + kb + 8);
          s = wmma_bf16(ak[ks].v, bq.v, s);
        }
        S[jt] = s;
      }

      // Online softmax (row stats per VGPR index; reduce across 16-lane halves)
#pragma unroll
      for (int r = 0; r < 8; ++r) {
        float sv0 = S[0][r] * sc;
        float sv1 = S[1][r] * sc;
        int trow = rbase + r;
        if (j0 + ln > trow) sv0 = -1e30f;
        if (j0 + 16 + ln > trow) sv1 = -1e30f;
        float rowm = fmaxf(sv0, sv1);
#pragma unroll
        for (int o = 1; o < 16; o <<= 1)
          rowm = fmaxf(rowm, __shfl_xor(rowm, o, 16));
        float mnew = fmaxf(mrow[r], rowm);
        float corr = exp2f(mrow[r] - mnew);
        float p0 = exp2f(sv0 - mnew);
        float p1 = exp2f(sv1 - mnew);
        float rs = p0 + p1;
#pragma unroll
        for (int o = 1; o < 16; o <<= 1) rs += __shfl_xor(rs, o, 16);
        lrow[r] = lrow[r] * corr + rs;
        mrow[r] = mnew;
#pragma unroll
        for (int nt = 0; nt < 4; ++nt) O[nt][r] *= corr;
        // Stage P bf16 into wave-private LDS (C-layout -> A-layout relayout)
        Ps[wave][(half * 8 + r) * 40 + ln] = (bf16)p0;
        Ps[wave][(half * 8 + r) * 40 + 16 + ln] = (bf16)p1;
      }

      // P A-fragment (16x32) and O += P @ V
      ABFrag ap;
      {
        int kb = half ? 16 : 0;
        ap.h[0] = *(const bf16x8*)(&Ps[wave][ln * 40 + kb]);
        ap.h[1] = *(const bf16x8*)(&Ps[wave][ln * 40 + kb + 8]);
      }
#pragma unroll
      for (int nt = 0; nt < 4; ++nt) {
        ABFrag bv;
        int nrow = nt * 16 + ln;
        int kb = half ? 16 : 0;
        bv.h[0] = *(const bf16x8*)(Vc + nrow * 40 + kb);
        bv.h[1] = *(const bf16x8*)(Vc + nrow * 40 + kb + 8);
        O[nt] = wmma_bf16(ap.v, bv.v, O[nt]);
      }
    }
  }

  // Normalize and store y[b][t][h*64+s'] (row-major 4096x768 for out-proj)
#pragma unroll
  for (int nt = 0; nt < 4; ++nt) {
    int ss = nt * 16 + ln;
#pragma unroll
    for (int r = 0; r < 8; ++r) {
      int t = rbase + r;
      float val = O[nt][r] / lrow[r];
      Y[((size_t)(b * T_DIM + t)) * C_DIM + h * HS_DIM + ss] = (bf16)val;
    }
  }
}

// ---------------------------------------------------------------------------
// Kernel 3: output projection. out = Y @ Wo + bo (fp32 out).
// grid = (N/64, M/128), block = 256. A tile (bf16) double-buffered via async
// DMA; W tile converted through VGPRs.
// ---------------------------------------------------------------------------
__global__ __launch_bounds__(256) void out_proj_kernel(
    const bf16* __restrict__ Yp, const float* __restrict__ Wo,
    const float* __restrict__ bo, float* __restrict__ out) {
  const int m0 = blockIdx.y * 128;
  const int n0 = blockIdx.x * 64;
  const int tid = threadIdx.x;
  const int lane = tid & 31;
  const int wave = tid >> 5;
  const int wm = wave & 3, wn = wave >> 2;
  const int ln = lane & 15, half = lane >> 4;

  __shared__ __align__(16) bf16 As[2][128 * 40];
  __shared__ __align__(16) bf16 Bs[64 * 40];

  const v8f vzero = {0.f, 0.f, 0.f, 0.f, 0.f, 0.f, 0.f, 0.f};
  v8f acc[2][2] = {{vzero, vzero}, {vzero, vzero}};

  // Per-thread async staging addresses (2 x b128 per thread per tile).
  const int ar[2] = {(tid * 8) >> 5, ((256 + tid) * 8) >> 5};
  const int ac[2] = {(tid * 8) & 31, ((256 + tid) * 8) & 31};
  unsigned a_dst[2][2];
#pragma unroll
  for (int buf = 0; buf < 2; ++buf)
#pragma unroll
    for (int p = 0; p < 2; ++p)
      a_dst[buf][p] = lds_off(As[buf] + ar[p] * 40 + ac[p]);

  // Prologue: kick off k-tile 0 into buffer 0.
#pragma unroll
  for (int p = 0; p < 2; ++p)
    async_copy_b128(a_dst[0][p], Yp + (size_t)(m0 + ar[p]) * C_DIM + ac[p]);

  const int nkt = C_DIM / 32;  // 24
  for (int kt = 0; kt < nkt; ++kt) {
    const int k0 = kt * 32;
    const int cur = kt & 1;
    __syncthreads();  // all waves done reading As[cur^1] and Bs
    // W tile: 4x4 patches, packed b64 stores into [n][k]
    if (tid < 128) {
      const int ng = (tid & 15) * 4;
      const int kg = (tid >> 4) * 4;
      const float* wp = Wo + (size_t)(k0 + kg) * C_DIM + n0 + ng;
      const float4 f0 = *(const float4*)(wp);
      const float4 f1 = *(const float4*)(wp + C_DIM);
      const float4 f2 = *(const float4*)(wp + 2 * C_DIM);
      const float4 f3 = *(const float4*)(wp + 3 * C_DIM);
      bf16x4 c0 = {(bf16)f0.x, (bf16)f1.x, (bf16)f2.x, (bf16)f3.x};
      bf16x4 c1 = {(bf16)f0.y, (bf16)f1.y, (bf16)f2.y, (bf16)f3.y};
      bf16x4 c2 = {(bf16)f0.z, (bf16)f1.z, (bf16)f2.z, (bf16)f3.z};
      bf16x4 c3 = {(bf16)f0.w, (bf16)f1.w, (bf16)f2.w, (bf16)f3.w};
      *(bf16x4*)(Bs + (ng + 0) * 40 + kg) = c0;
      *(bf16x4*)(Bs + (ng + 1) * 40 + kg) = c1;
      *(bf16x4*)(Bs + (ng + 2) * 40 + kg) = c2;
      *(bf16x4*)(Bs + (ng + 3) * 40 + kg) = c3;
    }
    if (kt + 1 < nkt) {
      const int kn = k0 + 32;
#pragma unroll
      for (int p = 0; p < 2; ++p)
        async_copy_b128(a_dst[cur ^ 1][p],
                        Yp + (size_t)(m0 + ar[p]) * C_DIM + kn + ac[p]);
      async_wait_le2();  // current A tile landed; next still in flight
    } else {
      async_wait0();
    }
    __syncthreads();

    const bf16* Ac = As[cur];
    ABFrag af[2], bfm[2];
#pragma unroll
    for (int mt = 0; mt < 2; ++mt) {
      int row = wm * 32 + mt * 16 + ln;
      int kb = half ? 16 : 0;
      af[mt].h[0] = *(const bf16x8*)(Ac + row * 40 + kb);
      af[mt].h[1] = *(const bf16x8*)(Ac + row * 40 + kb + 8);
    }
#pragma unroll
    for (int nt = 0; nt < 2; ++nt) {
      int nrow = wn * 32 + nt * 16 + ln;
      int kb = half ? 16 : 0;
      bfm[nt].h[0] = *(const bf16x8*)(Bs + nrow * 40 + kb);
      bfm[nt].h[1] = *(const bf16x8*)(Bs + nrow * 40 + kb + 8);
    }
#pragma unroll
    for (int mt = 0; mt < 2; ++mt)
#pragma unroll
      for (int nt = 0; nt < 2; ++nt)
        acc[mt][nt] = wmma_bf16(af[mt].v, bfm[nt].v, acc[mt][nt]);
  }

#pragma unroll
  for (int nt = 0; nt < 2; ++nt) {
    int n = n0 + wn * 32 + nt * 16 + ln;
    float bias = bo[n];
#pragma unroll
    for (int mt = 0; mt < 2; ++mt) {
#pragma unroll
      for (int r = 0; r < 8; ++r) {
        int m = m0 + wm * 32 + mt * 16 + half * 8 + r;
        out[(size_t)m * C_DIM + n] = acc[mt][nt][r] + bias;
      }
    }
  }
}

// ---------------------------------------------------------------------------
extern "C" void kernel_launch(void* const* d_in, const int* in_sizes, int n_in,
                              void* d_out, int out_size, void* d_ws,
                              size_t ws_size, hipStream_t stream) {
  (void)in_sizes;
  (void)n_in;
  (void)out_size;
  (void)ws_size;
  const float* x = (const float*)d_in[0];
  const float* Wk = (const float*)d_in[1];
  const float* bk = (const float*)d_in[2];
  const float* Wq = (const float*)d_in[3];
  const float* bq = (const float*)d_in[4];
  const float* Wv = (const float*)d_in[5];
  const float* bv = (const float*)d_in[6];
  const float* Wo = (const float*)d_in[7];
  const float* bo = (const float*)d_in[8];
  float* out = (float*)d_out;

  const size_t elems = (size_t)B_DIM * H_DIM * T_DIM * HS_DIM;  // 3,145,728
  bf16* Kh = (bf16*)d_ws;
  bf16* Qh = Kh + elems;
  bf16* Vt = Qh + elems;
  bf16* Y = Vt + elems;

  dim3 blk(256);
  qkv_proj_kernel<<<dim3(C_DIM / 64, M_DIM / 128, 3), blk, 0, stream>>>(
      x, Wk, bk, Wq, bq, Wv, bv, Kh, Qh, Vt);
  flash_attn_kernel<<<dim3(T_DIM / 128, H_DIM, B_DIM), blk, 0, stream>>>(
      Kh, Qh, Vt, Y);
  out_proj_kernel<<<dim3(C_DIM / 64, M_DIM / 128), blk, 0, stream>>>(Y, Wo, bo,
                                                                     out);
}
